// MTL2CharCNNWordBilstmModel_32796370273076
// MI455X (gfx1250) — compile-verified, matrix operand order checked by reading, not run on hardware
//
#include <hip/hip_runtime.h>
#include <hip/hip_bf16.h>

// ---------------------------------------------------------------------------
// MI455X (gfx1250, wave32) implementation of a 3x BiLSTM + 2x CRF model.
// B=64, T=512, D=300 (padded 320), H=256, 4H=1024, K1=10, K2=12.
//
// Strategy: pull the time-parallel x@Wx projection out of the recurrence as a
// big bf16 WMMA GEMM (Pd); run the sequential h@Wh recurrence as one
// workgroup per direction with h resident in LDS (WMMA-A swizzled bf16) and
// cell state c entirely in registers. All matrix math uses
// v_wmma_f32_16x16x32_bf16.
// ---------------------------------------------------------------------------

typedef __attribute__((ext_vector_type(16))) __bf16 v16bf;
typedef __attribute__((ext_vector_type(8)))  float  v8f;

#define B_   64
#define T_   512
#define D_   300
#define DP_  320        // D padded to multiple of 32
#define H_   256
#define G4_  1024       // 4*H
#define KT_X 10         // DP_/32 k-tiles for x projection
#define KT_H 8          // H_/32 k-tiles for recurrence
#define NT_  64         // G4_/16 n-tiles
#define MT_  2048       // (B_*T_)/16 m-tiles (time-major rows r = t*B + b)

__device__ __forceinline__ __bf16 f2bf(float f) {
    unsigned u = __builtin_bit_cast(unsigned, f);
    unsigned r = (u + 0x7FFFu + ((u >> 16) & 1u)) >> 16;   // RNE
    return __builtin_bit_cast(__bf16, (unsigned short)r);
}
__device__ __forceinline__ float sigm(float x) { return 1.0f / (1.0f + __expf(-x)); }

struct Ptrs6 { const float* p[6]; };

// ---------------------------------------------------------------------------
// Kernel 1: X (B,T,D) f32 -> bf16 A-tiles, rows r = t*B + b (time-major),
// K padded to 320 with zeros. Writes both forward order (Xf) and per-batch
// length-reversed order (Xr) for the backward LSTMs.
// A-tile swizzle (16-bit A 16x32 per ISA): lane = m + 16*half,
// element e in lane covers k = ((e>>3)<<4) | (half<<3) | (e&7).
// ---------------------------------------------------------------------------
__global__ __launch_bounds__(256) void prep_x(const float* __restrict__ x,
                                              const int* __restrict__ lengths,
                                              __bf16* __restrict__ Xf,
                                              __bf16* __restrict__ Xr) {
    size_t i = (size_t)blockIdx.x * 256 + threadIdx.x;      // MT_*KT_X*512 elems
    if (i >= (size_t)MT_ * KT_X * 512) return;
    int within = (int)(i & 511);
    size_t tile = i >> 9;
    int kt = (int)(tile % KT_X);
    int mt = (int)(tile / KT_X);
    int lane = within >> 4, e = within & 15;
    int m = lane & 15, half = lane >> 4;
    int r = mt * 16 + m;
    int t = r >> 6, b = r & 63;                             // r = t*64 + b
    int kk = ((e >> 3) << 4) | (half << 3) | (e & 7);
    int col = kt * 32 + kk;
    float vf = 0.f, vr = 0.f;
    if (col < D_) {
        vf = x[((size_t)b * T_ + t) * D_ + col];
        int len = lengths[b];
        int ts = len - 1 - t; if (ts < 0) ts = t;           // reference reverse_seq
        vr = x[((size_t)b * T_ + ts) * D_ + col];
    }
    Xf[i] = f2bf(vf);
    Xr[i] = f2bf(vr);
}

// ---------------------------------------------------------------------------
// Kernel 2: split each W (556,1024) into Wx (K=320 padded) and Wh (K=256),
// bf16, pre-swizzled into WMMA B-tile lane layout:
// lane = n + 16*half, element e covers k = ((e>>3)<<4) | (half<<3) | (e&7).
// Tile order: (kt*64 + nt).
// ---------------------------------------------------------------------------
#define WX_ELEMS (KT_X * NT_ * 512)    // 327680 per dir
#define WH_ELEMS (KT_H * NT_ * 512)    // 262144 per dir
__global__ __launch_bounds__(256) void prep_w(Ptrs6 Ws,
                                              __bf16* __restrict__ WxAll,
                                              __bf16* __restrict__ WhAll) {
    size_t i = (size_t)blockIdx.x * 256 + threadIdx.x;
    if (i >= (size_t)6 * (WX_ELEMS + WH_ELEMS)) return;
    int dir = (int)(i / (WX_ELEMS + WH_ELEMS));
    int rem = (int)(i % (WX_ELEMS + WH_ELEMS));
    bool isX = rem < WX_ELEMS;
    int r2 = isX ? rem : (rem - WX_ELEMS);
    int tile = r2 >> 9, within = r2 & 511;
    int kt = tile / NT_, nt = tile % NT_;
    int lane = within >> 4, e = within & 15;
    int nq = lane & 15, half = lane >> 4;
    int kk = ((e >> 3) << 4) | (half << 3) | (e & 7);
    int k = kt * 32 + kk;
    const float* W = Ws.p[dir];                             // (556, 1024) row-major
    float v = 0.f;
    if (isX) { if (k < D_) v = W[(size_t)k * G4_ + nt * 16 + nq]; }
    else     { v = W[(size_t)(D_ + k) * G4_ + nt * 16 + nq]; }
    if (isX) WxAll[(size_t)dir * WX_ELEMS + r2] = f2bf(v);
    else     WhAll[(size_t)dir * WH_ELEMS + r2] = f2bf(v);
}

// ---------------------------------------------------------------------------
// Kernel 3: Pd[dir] = X(bf16) @ Wx(bf16) + bias, f32, stored D-tile-swizzled:
// tile (mt*64+nt), per lane 8 consecutive floats (lane*8 + v), so the
// recurrent kernel loads it as contiguous v8f. Grid (2048, 8, 6), 8 waves.
// ---------------------------------------------------------------------------
__global__ __launch_bounds__(256) void xproj(const __bf16* __restrict__ Xf,
                                             const __bf16* __restrict__ Xr,
                                             const __bf16* __restrict__ WxAll,
                                             Ptrs6 biases,
                                             float* __restrict__ PdAll) {
    int mt  = blockIdx.x;
    int nt  = blockIdx.y * 8 + (threadIdx.x >> 5);
    int dir = blockIdx.z;
    int lane = threadIdx.x & 31;
    const __bf16* A  = (dir & 1) ? Xr : Xf;
    const __bf16* Bm = WxAll + (size_t)dir * WX_ELEMS;
    v8f acc = {0.f, 0.f, 0.f, 0.f, 0.f, 0.f, 0.f, 0.f};
#pragma unroll
    for (int k = 0; k < KT_X; ++k) {
        v16bf a  = *(const v16bf*)&A [((size_t)mt * KT_X + k) * 512 + lane * 16];
        v16bf bt = *(const v16bf*)&Bm[((size_t)k  * NT_  + nt) * 512 + lane * 16];
        acc = __builtin_amdgcn_wmma_f32_16x16x32_bf16(false, a, false, bt,
                                                      (short)0, acc, false, false);
    }
    float bias = biases.p[dir][nt * 16 + (lane & 15)];
#pragma unroll
    for (int v = 0; v < 8; ++v) acc[v] += bias;
    *(v8f*)&PdAll[(size_t)dir * ((size_t)MT_ * NT_ * 256)
                  + ((size_t)mt * NT_ + nt) * 256 + lane * 8] = acc;
}

// ---------------------------------------------------------------------------
// Kernel 4: the recurrence. One block (32 waves) per direction.
// Wave w: m-tile mt = w&3 (16 batch rows), hidden group hg = w>>2; it owns
// hidden tiles {hg, hg+8} across all 4 gates -> i,j,f,o combine in-register.
// h lives in LDS in A-swizzled bf16; c lives in registers (mapping is
// step-invariant). bf16 WMMA h@Wh each step; Pd + bias added in epilogue.
// ---------------------------------------------------------------------------
__global__ __launch_bounds__(1024) void lstm_rec(const __bf16* __restrict__ WhAll,
                                                 const float* __restrict__ PdAll,
                                                 const int* __restrict__ lengths,
                                                 float* __restrict__ hOutAll) {
    __shared__ __bf16 hA[4 * KT_H * 512];                   // 32 KB, A-swizzled h
    int dir = blockIdx.x;
    const __bf16* Wh = WhAll + (size_t)dir * WH_ELEMS;
    const float*  Pd = PdAll + (size_t)dir * ((size_t)MT_ * NT_ * 256);
    float* hOut = hOutAll + (size_t)dir * ((size_t)B_ * T_ * H_);
    const bool bw = dir & 1;

    int tid = threadIdx.x, lane = tid & 31, wv = tid >> 5;
    int mt = wv & 3, hg = wv >> 2;
    int nq = lane & 15, half = lane >> 4;

    for (int i = tid; i < 4 * KT_H * 512; i += 1024)
        hA[i] = __builtin_bit_cast(__bf16, (unsigned short)0);

    // Step-invariant per-element state: batch row, length, cell state.
    int   bidx[8]; int lenV[8]; float cReg[2][8];
#pragma unroll
    for (int v = 0; v < 8; ++v) {
        bidx[v] = mt * 16 + v + 8 * half;
        lenV[v] = lengths[bidx[v]];
        cReg[0][v] = 0.f; cReg[1][v] = 0.f;
    }
    __syncthreads();

    for (int t = 0; t < T_; ++t) {
        v8f acc[2][4];
#pragma unroll
        for (int hp = 0; hp < 2; ++hp)
#pragma unroll
            for (int g = 0; g < 4; ++g)
                acc[hp][g] = (v8f){0.f, 0.f, 0.f, 0.f, 0.f, 0.f, 0.f, 0.f};

#pragma unroll
        for (int k = 0; k < KT_H; ++k) {
            v16bf a = *(const v16bf*)&hA[((mt * KT_H) + k) * 512 + lane * 16];
#pragma unroll
            for (int hp = 0; hp < 2; ++hp) {
                int ht = hg + hp * 8;
#pragma unroll
                for (int g = 0; g < 4; ++g) {
                    int nt = g * 16 + ht;
                    v16bf bt = *(const v16bf*)&Wh[((size_t)k * NT_ + nt) * 512 + lane * 16];
                    acc[hp][g] = __builtin_amdgcn_wmma_f32_16x16x32_bf16(
                        false, a, false, bt, (short)0, acc[hp][g], false, false);
                }
            }
        }
        __syncthreads();   // all waves done reading hA before anyone rewrites it

#pragma unroll
        for (int hp = 0; hp < 2; ++hp) {
            int ht = hg + hp * 8;
            int hid = ht * 16 + nq;
            v8f pz[4];
#pragma unroll
            for (int g = 0; g < 4; ++g)
                pz[g] = *(const v8f*)&Pd[(((size_t)t * 4 + mt) * NT_ + (g * 16 + ht)) * 256
                                         + lane * 8];
#pragma unroll
            for (int v = 0; v < 8; ++v) {
                float zi = acc[hp][0][v] + pz[0][v];
                float zj = acc[hp][1][v] + pz[1][v];
                float zf = acc[hp][2][v] + pz[2][v];
                float zo = acc[hp][3][v] + pz[3][v];
                float cn = sigm(zf + 1.0f) * cReg[hp][v] + sigm(zi) * tanhf(zj);
                float hn = sigm(zo) * tanhf(cn);
                bool mk = t < lenV[v];
                int tdst = t;
                if (bw) { int ts = lenV[v] - 1 - t; tdst = (ts >= 0) ? ts : t; }
                hOut[((size_t)bidx[v] * T_ + tdst) * H_ + hid] = mk ? hn : 0.f;
                if (mk) {
                    cReg[hp][v] = cn;
                    int kk = hid & 31, kt2 = hid >> 5;
                    int h2 = (kk >> 3) & 1;
                    int e  = ((kk >> 4) << 3) | (kk & 7);
                    hA[((mt * KT_H) + kt2) * 512 + ((bidx[v] & 15) + 16 * h2) * 16 + e]
                        = f2bf(hn);
                }
            }
        }
        __syncthreads();   // hA fully updated before next step's reads
    }
}

// ---------------------------------------------------------------------------
// Kernel 5: logits1 = [t1fw|t1bw|shfw|shbw] @ W1 + b1 (K1=10),
//           logits2 = [t2fw|t2bw|shfw|shbw] @ W2 + b2 (K2=12).
// N is tiny -> plain VALU dot products, one thread per (b,t).
// ---------------------------------------------------------------------------
__global__ __launch_bounds__(256) void proj(const float* __restrict__ hOutAll,
                                            const float* __restrict__ W1,
                                            const float* __restrict__ b1,
                                            const float* __restrict__ W2,
                                            const float* __restrict__ b2,
                                            float* __restrict__ lg1,
                                            float* __restrict__ lg2) {
    int i = blockIdx.x * 256 + threadIdx.x;                 // (b*T + t)
    if (i >= B_ * T_) return;
    const int d1[4] = {0, 1, 4, 5}, d2[4] = {2, 3, 4, 5};
    float a1[10], a2[12];
#pragma unroll
    for (int k = 0; k < 10; ++k) a1[k] = b1[k];
#pragma unroll
    for (int k = 0; k < 12; ++k) a2[k] = b2[k];
    for (int seg = 0; seg < 4; ++seg) {
        const float* f1 = hOutAll + ((size_t)d1[seg] * B_ * T_ + i) * H_;
        const float* f2 = hOutAll + ((size_t)d2[seg] * B_ * T_ + i) * H_;
        int hbase = seg * H_;
        for (int hh = 0; hh < H_; ++hh) {
            float v1 = f1[hh], v2 = f2[hh];
            const float* w1r = W1 + (size_t)(hbase + hh) * 10;
            const float* w2r = W2 + (size_t)(hbase + hh) * 12;
#pragma unroll
            for (int k = 0; k < 10; ++k) a1[k] += v1 * w1r[k];
#pragma unroll
            for (int k = 0; k < 12; ++k) a2[k] += v2 * w2r[k];
        }
    }
#pragma unroll
    for (int k = 0; k < 10; ++k) lg1[(size_t)i * 10 + k] = a1[k];
#pragma unroll
    for (int k = 0; k < 12; ++k) lg2[(size_t)i * 12 + k] = a2[k];
}

// ---------------------------------------------------------------------------
// Kernel 6: CRF NLL. One wave per (batch, task); lane j holds alpha_j,
// cross-lane gather via __shfl (wave32). nll = logZ - (unary + trans).
// ---------------------------------------------------------------------------
__global__ __launch_bounds__(32) void crf(const float* __restrict__ lg1,
                                          const float* __restrict__ lg2,
                                          const int* __restrict__ labels,
                                          const int* __restrict__ lengths,
                                          const float* __restrict__ tr1,
                                          const float* __restrict__ tr2,
                                          float* __restrict__ out) {
    int b = blockIdx.x, task = blockIdx.y;
    int K = task ? 12 : 10;
    const float* lg = task ? lg2 : lg1;
    const float* tr = task ? tr2 : tr1;
    int lane = threadIdx.x;
    int len = lengths[b];
    const int* lab = labels + (size_t)b * T_;
    const float* lrow = lg + (size_t)b * T_ * K;
    const float NEG = -1e30f;

    float alpha = (lane < K) ? lrow[lane] : NEG;
    for (int t = 1; t < T_; ++t) {
        float av[12];
#pragma unroll
        for (int i2 = 0; i2 < 12; ++i2) av[i2] = __shfl(alpha, i2, 32);
        float m = NEG, s = 0.f, na = NEG;
        if (lane < K) {
            for (int i2 = 0; i2 < K; ++i2) m = fmaxf(m, av[i2] + tr[i2 * K + lane]);
            for (int i2 = 0; i2 < K; ++i2) s += __expf(av[i2] + tr[i2 * K + lane] - m);
            na = m + __logf(s) + lrow[(size_t)t * K + lane];
        }
        if (t < len && lane < K) alpha = na;
    }
    float m = alpha;
#pragma unroll
    for (int off = 16; off > 0; off >>= 1) m = fmaxf(m, __shfl_xor(m, off, 32));
    float s = (lane < K) ? __expf(alpha - m) : 0.f;
#pragma unroll
    for (int off = 16; off > 0; off >>= 1) s += __shfl_xor(s, off, 32);
    float logZ = m + __logf(s);

    float us = 0.f, ts = 0.f;
    for (int t = lane; t < T_; t += 32) {
        if (t < len) {
            us += lrow[(size_t)t * K + lab[t]];
            if (t >= 1) ts += tr[lab[t - 1] * K + lab[t]];
        }
    }
#pragma unroll
    for (int off = 16; off > 0; off >>= 1) {
        us += __shfl_xor(us, off, 32);
        ts += __shfl_xor(ts, off, 32);
    }
    if (lane == 0) out[task * B_ + b] = logZ - (us + ts);
}

// ---------------------------------------------------------------------------
extern "C" void kernel_launch(void* const* d_in, const int* in_sizes, int n_in,
                              void* d_out, int out_size, void* d_ws, size_t ws_size,
                              hipStream_t stream) {
    (void)in_sizes; (void)n_in; (void)out_size; (void)ws_size;
    const float* x       = (const float*)d_in[0];
    const int*   lengths = (const int*)  d_in[1];
    const int*   labels  = (const int*)  d_in[2];
    Ptrs6 Ws, Bs;
    for (int d = 0; d < 6; ++d) {
        Ws.p[d] = (const float*)d_in[3 + 2 * d];
        Bs.p[d] = (const float*)d_in[4 + 2 * d];
    }
    const float* W1  = (const float*)d_in[15];
    const float* b1  = (const float*)d_in[16];
    const float* W2  = (const float*)d_in[17];
    const float* b2  = (const float*)d_in[18];
    const float* tr1 = (const float*)d_in[19];
    const float* tr2 = (const float*)d_in[20];

    char* ws = (char*)d_ws;
    size_t off = 0;
    __bf16* Xf    = (__bf16*)(ws + off); off += (size_t)MT_ * KT_X * 512 * 2;   // 20.97 MB
    __bf16* Xr    = (__bf16*)(ws + off); off += (size_t)MT_ * KT_X * 512 * 2;   // 20.97 MB
    __bf16* WxAll = (__bf16*)(ws + off); off += (size_t)6 * WX_ELEMS * 2;       //  3.93 MB
    __bf16* WhAll = (__bf16*)(ws + off); off += (size_t)6 * WH_ELEMS * 2;       //  3.15 MB
    float*  PdAll = (float*) (ws + off); off += (size_t)6 * MT_ * NT_ * 256 * 4;// 805  MB
    float*  hOut  = (float*) (ws + off); off += (size_t)6 * B_ * T_ * H_ * 4;   // 201  MB
    float*  lg1   = (float*) (ws + off); off += (size_t)B_ * T_ * 10 * 4;
    float*  lg2   = (float*) (ws + off); off += (size_t)B_ * T_ * 12 * 4;

    {   // 1. X -> swizzled bf16 A-tiles (forward + reversed)
        size_t n = (size_t)MT_ * KT_X * 512;
        prep_x<<<dim3((unsigned)((n + 255) / 256)), dim3(256), 0, stream>>>(x, lengths, Xf, Xr);
    }
    {   // 2. W -> swizzled bf16 B-tiles (Wx + Wh)
        size_t n = (size_t)6 * (WX_ELEMS + WH_ELEMS);
        prep_w<<<dim3((unsigned)((n + 255) / 256)), dim3(256), 0, stream>>>(Ws, WxAll, WhAll);
    }
    // 3. Pd = X @ Wx + bias  (bulk WMMA GEMM)
    xproj<<<dim3(MT_, 8, 6), dim3(256), 0, stream>>>(Xf, Xr, WxAll, Bs, PdAll);
    // 4. Sequential LSTM recurrence (WMMA per step, 6 directions)
    lstm_rec<<<dim3(6), dim3(1024), 0, stream>>>(WhAll, PdAll, lengths, hOut);
    // 5. Output projections (tiny N)
    proj<<<dim3((B_ * T_ + 255) / 256), dim3(256), 0, stream>>>(hOut, W1, b1, W2, b2, lg1, lg2);
    // 6. CRF negative log-likelihood -> d_out (2, B)
    crf<<<dim3(B_, 2), dim3(32), 0, stream>>>(lg1, lg2, labels, lengths, tr1, tr2, (float*)d_out);
}
